// GCN_73778948211120
// MI455X (gfx1250) — matrix operand order repaired and checked by reading, not compiled, not run on hardware
//
// GCN 3-layer forward for MI455X (gfx1250), compile-only target.
//
// All heavy math (two T*diag(s)*T^T node products, one T^T*diag(s)*T edge
// product, and the A @ (H W) aggregations, ~155 GFLOP total) runs on
// v_wmma_f32_16x16x32_bf16 (bf16 in, fp32 accumulate). Inputs are cast to
// bf16 once; elementwise adj/diag masking is fused between the two WMMA
// chains of each node layer via an in-register -> LDS -> A-fragment
// transpose. Edge-layer column-max division is folded into the HW rows so
// pass 2 is a pure WMMA GEMM over the stored bf16 A matrix.

#include <hip/hip_runtime.h>
#include <float.h>

typedef __bf16 bf16;
typedef __attribute__((ext_vector_type(16))) __bf16 v16bf;
typedef __attribute__((ext_vector_type(8)))  __bf16 v8bf;
typedef __attribute__((ext_vector_type(8)))  float  v8f;

constexpr int B_ = 8, NV_ = 1024, NE_ = 2048;
constexpr int FV_ = 256, FE_ = 128, NH_ = 256, FMID_ = 128;

// ---------------- fragment helpers (wave32 WMMA layouts) ----------------

__device__ __forceinline__ v8f vzero8() {
  v8f v;
#pragma unroll
  for (int i = 0; i < 8; ++i) v[i] = 0.0f;
  return v;
}

// A operand (16x32 bf16): lane m%16 = row, lane>=16 takes K 8..15 / 24..31.
// Per lane: 8 bf16 at p, 8 bf16 at p+16 (p already offset by (lane>>4)*8).
__device__ __forceinline__ v16bf load_frag_a(const bf16* p) {
  v8bf lo = *(const v8bf*)p;
  v8bf hi = *(const v8bf*)(p + 16);
  v16bf r;
#pragma unroll
  for (int i = 0; i < 8; ++i) { r[i] = lo[i]; r[i + 8] = hi[i]; }
  return r;
}

// B operand (32x16 bf16): lane%16 = column n, lane>=16 takes K 8..15 / 24..31.
// Per lane: 8 bf16 at p, 8 bf16 at p+16 (memory stored [n][k], p offset by (lane>>4)*8).
__device__ __forceinline__ v16bf load_frag_b(const bf16* p) {
  v8bf lo = *(const v8bf*)p;
  v8bf hi = *(const v8bf*)(p + 16);
  v16bf r;
#pragma unroll
  for (int i = 0; i < 8; ++i) { r[i] = lo[i]; r[i + 8] = hi[i]; }
  return r;
}

__device__ __forceinline__ v8f wmma_bf16(v16bf a, v16bf b, v8f c) {
  return __builtin_amdgcn_wmma_f32_16x16x32_bf16(false, a, false, b,
                                                 (short)0, c, false, false);
}

__device__ __forceinline__ void atomicMaxF(float* addr, float v) {
  if (v >= 0.0f) atomicMax((int*)addr, __float_as_int(v));
  else           atomicMin((unsigned int*)addr, __float_as_uint(v));
}

// ---------------- small elementwise / cast kernels ----------------

__global__ void cast_bf16_kernel(const float* __restrict__ src,
                                 bf16* __restrict__ dst, long n) {
  long i = (long)blockIdx.x * blockDim.x + threadIdx.x;
  if (i < n) dst[i] = (bf16)src[i];
}

// dst[row, c] = bf16(src[row, c] * s[b, c]),  b = row / rowsPerB, L = row len
__global__ void scale_cast_f32_kernel(const float* __restrict__ src,
                                      const float* __restrict__ s,
                                      bf16* __restrict__ dst,
                                      int rowsPerB, int L, long n) {
  long i = (long)blockIdx.x * blockDim.x + threadIdx.x;
  if (i >= n) return;
  long row = i / L;
  int  c   = (int)(i - row * L);
  int  b   = (int)(row / rowsPerB);
  dst[i] = (bf16)(src[i] * s[(size_t)b * L + c]);
}

__global__ void scale_cast_bf16_kernel(const bf16* __restrict__ src,
                                       const float* __restrict__ s,
                                       bf16* __restrict__ dst,
                                       int rowsPerB, int L, long n) {
  long i = (long)blockIdx.x * blockDim.x + threadIdx.x;
  if (i >= n) return;
  long row = i / L;
  int  c   = (int)(i - row * L);
  int  b   = (int)(row / rowsPerB);
  dst[i] = (bf16)((float)src[i] * s[(size_t)b * L + c]);
}

// Tt[b, e, i] = bf16(T[b, i, e])   (N1 = NV, N2 = NE)
__global__ void transpose_cast_kernel(const float* __restrict__ T,
                                      bf16* __restrict__ Tt,
                                      int N1, int N2, long n) {
  long idx = (long)blockIdx.x * blockDim.x + threadIdx.x;
  if (idx >= n) return;
  int  i = (int)(idx % N1);
  long t = idx / N1;
  int  e = (int)(t % N2);
  int  b = (int)(t / N2);
  Tt[idx] = (bf16)T[((size_t)b * N1 + i) * N2 + e];
}

// WT[k, f] = bf16(W[f, k])
__global__ void wtrans_cast_kernel(const float* __restrict__ W,
                                   bf16* __restrict__ WT, int Fin, int Fout) {
  int idx = blockIdx.x * blockDim.x + threadIdx.x;
  if (idx >= Fin * Fout) return;
  int f = idx % Fin, k = idx / Fin;
  WT[idx] = (bf16)W[(size_t)f * Fout + k];
}

__global__ void fill_kernel(float* __restrict__ p, float v, int n) {
  int i = blockIdx.x * blockDim.x + threadIdx.x;
  if (i < n) p[i] = v;
}

// s[row] = dot(H[row, :], p)   one wave per row
__global__ __launch_bounds__(128) void rowdot_kernel(
    const float* __restrict__ H, const float* __restrict__ p,
    float* __restrict__ s, int rows, int F) {
  int row = blockIdx.x * 4 + threadIdx.y;
  if (row >= rows) return;
  int lane = threadIdx.x;
  const float* h = H + (size_t)row * F;
  float acc = 0.0f;
  for (int f = lane; f < F; f += 32) acc += h[f] * p[f];
#pragma unroll
  for (int off = 16; off > 0; off >>= 1) acc += __shfl_xor(acc, off, 32);
  if (lane == 0) s[row] = acc;
}

// ---------------- dense WMMA GEMM: HWT[b,k,j] = scale_j * sum_f H[b,j,f] WT[k,f]
// Output stored transposed (K-major) so it can be consumed as a B operand.

__global__ __launch_bounds__(32) void gemm_hwt_kernel(
    const bf16* __restrict__ Hbf,  // [B, NJ, Fin]
    const bf16* __restrict__ WT,   // [Fout, Fin]
    const float* __restrict__ colmax,  // [B, NJ] or null -> scale = 1/colmax
    bf16* __restrict__ HWT,        // [B, Fout, NJ]
    int NJ, int Fin, int Fout) {
  const int b = blockIdx.z;
  const int j0 = blockIdx.x * 16, k0 = blockIdx.y * 16;
  const int lane = threadIdx.x, ln = lane & 15, lh = lane >> 4;
  const bf16* aB = Hbf + ((size_t)b * NJ + j0 + ln) * Fin + lh * 8;
  const bf16* bB = WT + (size_t)(k0 + ln) * Fin + lh * 8;
  v8f acc = vzero8();
  for (int f = 0; f < Fin; f += 32)
    acc = wmma_bf16(load_frag_a(aB + f), load_frag_b(bB + f), acc);
#pragma unroll
  for (int r = 0; r < 8; ++r) {
    int jj = j0 + r + lh * 8;
    float v = acc[r];
    if (colmax) v *= 1.0f / colmax[(size_t)b * NJ + jj];
    HWT[((size_t)b * Fout + k0 + ln) * NJ + jj] = (bf16)v;
  }
}

// ---------------- fused node layer: out = ((eye + (1-eye)*Ts@T^T) * adj) @ HW + b

template <int F>
__global__ __launch_bounds__(32) void node_fused_kernel(
    const bf16* __restrict__ Ts,   // [B, NV, NE]  A operand (T * s)
    const bf16* __restrict__ Tb,   // [B, NV, NE]  B operand rows (T)
    const float* __restrict__ adj, // [B, NV, NV]
    const bf16* __restrict__ HWT,  // [B, F, NV]   (Hv @ W)^T
    const float* __restrict__ bias,
    float* __restrict__ out,       // [B, NV, F]
    bf16* __restrict__ out_bf,     // optional bf16 mirror
    int relu) {
  constexpr int NT = F / 16;
  __shared__ __attribute__((aligned(16))) bf16 a2lds[16 * 32];
  const int b = blockIdx.y;
  const int i0 = blockIdx.x * 16;
  const int lane = threadIdx.x, ln = lane & 15, lh = lane >> 4;

  const bf16* aBase = Ts + ((size_t)b * NV_ + i0 + ln) * NE_ + lh * 8;
  const bf16* TbB = Tb + (size_t)b * NV_ * NE_;
  const float* adjB = adj + ((size_t)b * NV_ + i0) * NV_;
  const bf16* hB = HWT + (size_t)b * F * NV_;

  v8f acc[NT];
#pragma unroll
  for (int t = 0; t < NT; ++t) acc[t] = vzero8();

  for (int j0 = 0; j0 < NV_; j0 += 32) {
    v8f m0 = vzero8(), m1 = vzero8();
    const bf16* b0 = TbB + (size_t)(j0 + ln) * NE_ + lh * 8;
    const bf16* b1 = b0 + (size_t)16 * NE_;
#pragma unroll 4
    for (int e = 0; e < NE_; e += 32) {
      v16bf a = load_frag_a(aBase + e);
      m0 = wmma_bf16(a, load_frag_b(b0 + e), m0);
      m1 = wmma_bf16(a, load_frag_b(b1 + e), m1);
    }
    // mask (diag=1) * adj, f32 -> bf16, C-layout -> A-layout via LDS
#pragma unroll
    for (int r = 0; r < 8; ++r) {
      const int mi = r + lh * 8;
      const float* arow = adjB + (size_t)mi * NV_ + j0;
      float v0 = (((i0 + mi) == (j0 + ln)) ? 1.0f : m0[r]) * arow[ln];
      float v1 = (((i0 + mi) == (j0 + 16 + ln)) ? 1.0f : m1[r]) * arow[16 + ln];
      a2lds[mi * 32 + ln] = (bf16)v0;
      a2lds[mi * 32 + 16 + ln] = (bf16)v1;
    }
    asm volatile("s_wait_dscnt 0x0" ::: "memory");  // intra-wave DS ordering
    v16bf a2 = load_frag_a(a2lds + ln * 32 + lh * 8);
#pragma unroll
    for (int t = 0; t < NT; ++t) {
      v16bf bb = load_frag_b(hB + (size_t)(t * 16 + ln) * NV_ + j0 + lh * 8);
      acc[t] = wmma_bf16(a2, bb, acc[t]);
    }
    asm volatile("s_wait_dscnt 0x0" ::: "memory");
  }
#pragma unroll
  for (int t = 0; t < NT; ++t) {
    const float bv = bias[t * 16 + ln];
#pragma unroll
    for (int r = 0; r < 8; ++r) {
      const int mi = r + lh * 8;
      float v = acc[t][r] + bv;
      if (relu) v = fmaxf(v, 0.0f);
      const size_t o = ((size_t)b * NV_ + i0 + mi) * F + t * 16 + ln;
      out[o] = v;
      if (out_bf) out_bf[o] = (bf16)v;
    }
  }
}

// ---------------- edge pass 1: A = ((eye + (1-eye)*Ts@Tt^T) * adj_e), colmax

__global__ __launch_bounds__(32) void edge_pass1_kernel(
    const bf16* __restrict__ TsE,  // [B, NE, NV]  (T^T * s2)
    const bf16* __restrict__ Ttb,  // [B, NE, NV]  T^T bf16 (B operand rows)
    const float* __restrict__ adjE,// [B, NE, NE]
    bf16* __restrict__ Aout,       // [B, NE, NE]
    float* __restrict__ colmax) {  // [B, NE], init -FLT_MAX
  const int b = blockIdx.y;
  const int i0 = blockIdx.x * 16;
  const int lane = threadIdx.x, ln = lane & 15, lh = lane >> 4;

  const bf16* aBase = TsE + ((size_t)b * NE_ + i0 + ln) * NV_ + lh * 8;
  const bf16* TtB = Ttb + (size_t)b * NE_ * NV_;
  const float* adjB = adjE + ((size_t)b * NE_ + i0) * NE_;
  bf16* AoutB = Aout + ((size_t)b * NE_ + i0) * NE_;
  float* cmB = colmax + (size_t)b * NE_;

  for (int j0 = 0; j0 < NE_; j0 += 32) {
    v8f m0 = vzero8(), m1 = vzero8();
    const bf16* b0 = TtB + (size_t)(j0 + ln) * NV_ + lh * 8;
    const bf16* b1 = b0 + (size_t)16 * NV_;
#pragma unroll 4
    for (int v = 0; v < NV_; v += 32) {
      v16bf a = load_frag_a(aBase + v);
      m0 = wmma_bf16(a, load_frag_b(b0 + v), m0);
      m1 = wmma_bf16(a, load_frag_b(b1 + v), m1);
    }
    float pm0 = -FLT_MAX, pm1 = -FLT_MAX;
#pragma unroll
    for (int r = 0; r < 8; ++r) {
      const int mi = r + lh * 8;
      const float* arow = adjB + (size_t)mi * NE_ + j0;
      float v0 = (((i0 + mi) == (j0 + ln)) ? 1.0f : m0[r]) * arow[ln];
      float v1 = (((i0 + mi) == (j0 + 16 + ln)) ? 1.0f : m1[r]) * arow[16 + ln];
      AoutB[(size_t)mi * NE_ + j0 + ln] = (bf16)v0;
      AoutB[(size_t)mi * NE_ + j0 + 16 + ln] = (bf16)v1;
      pm0 = fmaxf(pm0, v0);
      pm1 = fmaxf(pm1, v1);
    }
    pm0 = fmaxf(pm0, __shfl_xor(pm0, 16, 32));
    pm1 = fmaxf(pm1, __shfl_xor(pm1, 16, 32));
    if (lh == 0) {
      atomicMaxF(&cmB[j0 + ln], pm0);
      atomicMaxF(&cmB[j0 + 16 + ln], pm1);
    }
  }
}

// ---------------- edge pass 2 / generic: out = A @ HWT^T + bias (relu opt)

template <int F>
__global__ __launch_bounds__(32) void amatmul_kernel(
    const bf16* __restrict__ A,   // [B, M, K] row-major (A operand)
    const bf16* __restrict__ HWT, // [B, F, K] (B operand, K-major)
    const float* __restrict__ bias,
    float* __restrict__ out,      // [B, M, F]
    int M, int K, int relu) {
  constexpr int NT = F / 16;
  const int b = blockIdx.y;
  const int i0 = blockIdx.x * 16;
  const int lane = threadIdx.x, ln = lane & 15, lh = lane >> 4;
  const bf16* aBase = A + ((size_t)b * M + i0 + ln) * K + lh * 8;
  const bf16* hB = HWT + (size_t)b * F * K;
  v8f acc[NT];
#pragma unroll
  for (int t = 0; t < NT; ++t) acc[t] = vzero8();
  for (int k0 = 0; k0 < K; k0 += 32) {
    v16bf a = load_frag_a(aBase + k0);
#pragma unroll
    for (int t = 0; t < NT; ++t) {
      v16bf bb = load_frag_b(hB + (size_t)(t * 16 + ln) * K + k0 + lh * 8);
      acc[t] = wmma_bf16(a, bb, acc[t]);
    }
  }
#pragma unroll
  for (int t = 0; t < NT; ++t) {
    const float bv = bias[t * 16 + ln];
#pragma unroll
    for (int r = 0; r < 8; ++r) {
      const int mi = r + lh * 8;
      float v = acc[t][r] + bv;
      if (relu) v = fmaxf(v, 0.0f);
      out[((size_t)b * M + i0 + mi) * F + t * 16 + ln] = v;
    }
  }
}

// ---------------- host orchestration ----------------

extern "C" void kernel_launch(void* const* d_in, const int* in_sizes, int n_in,
                              void* d_out, int out_size, void* d_ws,
                              size_t ws_size, hipStream_t stream) {
  (void)in_sizes; (void)n_in; (void)out_size; (void)ws_size;
  const float* X     = (const float*)d_in[0];
  const float* Z     = (const float*)d_in[1];
  const float* adj_e = (const float*)d_in[2];
  const float* adj_v = (const float*)d_in[3];
  const float* T     = (const float*)d_in[4];
  const float* W1 = (const float*)d_in[5];
  const float* p1 = (const float*)d_in[6];
  const float* b1 = (const float*)d_in[7];
  const float* W2 = (const float*)d_in[8];
  const float* p2 = (const float*)d_in[9];
  const float* b2 = (const float*)d_in[10];
  const float* W3 = (const float*)d_in[11];
  const float* p3 = (const float*)d_in[12];
  const float* b3 = (const float*)d_in[13];
  float* out = (float*)d_out;

  char* w = (char*)d_ws;
  auto take = [&](size_t bytes) -> char* {
    char* p = w;
    w += (bytes + 255) & ~(size_t)255;
    return p;
  };
  bf16* Tbf   = (bf16*)take((size_t)B_ * NV_ * NE_ * 2);
  bf16* Ttbf  = (bf16*)take((size_t)B_ * NE_ * NV_ * 2);
  bf16* Ts    = (bf16*)take((size_t)B_ * NV_ * NE_ * 2);  // reused 3x
  bf16* Aedge = (bf16*)take((size_t)B_ * NE_ * NE_ * 2);
  bf16* Xbf   = (bf16*)take((size_t)B_ * NV_ * FV_ * 2);
  bf16* Zbf   = (bf16*)take((size_t)B_ * NE_ * FE_ * 2);
  bf16* X1bf  = (bf16*)take((size_t)B_ * NV_ * NH_ * 2);
  bf16* HWT1  = (bf16*)take((size_t)B_ * NH_ * NV_ * 2);
  bf16* HWT2  = (bf16*)take((size_t)B_ * FMID_ * NE_ * 2);
  bf16* HWT3  = (bf16*)take((size_t)B_ * FV_ * NV_ * 2);
  bf16* W1T   = (bf16*)take((size_t)FV_ * NH_ * 2);
  bf16* W2T   = (bf16*)take((size_t)FE_ * FMID_ * 2);
  bf16* W3T   = (bf16*)take((size_t)NH_ * FV_ * 2);
  float* X1   = (float*)take((size_t)B_ * NV_ * NH_ * 4);
  float* Z1   = (float*)take((size_t)B_ * NE_ * FMID_ * 4);
  float* s1   = (float*)take((size_t)B_ * NE_ * 4);
  float* s2   = (float*)take((size_t)B_ * NV_ * 4);
  float* s3   = (float*)take((size_t)B_ * NE_ * 4);
  float* cmax = (float*)take((size_t)B_ * NE_ * 4);

  auto blocks = [](long n) { return (unsigned)((n + 255) / 256); };

  // ---- one-time casts ----
  { long n = (long)B_ * NV_ * FV_;
    cast_bf16_kernel<<<blocks(n), 256, 0, stream>>>(X, Xbf, n); }
  { long n = (long)B_ * NE_ * FE_;
    cast_bf16_kernel<<<blocks(n), 256, 0, stream>>>(Z, Zbf, n); }
  { long n = (long)B_ * NV_ * NE_;
    cast_bf16_kernel<<<blocks(n), 256, 0, stream>>>(T, Tbf, n); }
  { long n = (long)B_ * NE_ * NV_;
    transpose_cast_kernel<<<blocks(n), 256, 0, stream>>>(T, Ttbf, NV_, NE_, n); }
  wtrans_cast_kernel<<<blocks((long)FV_ * NH_), 256, 0, stream>>>(W1, W1T, FV_, NH_);
  wtrans_cast_kernel<<<blocks((long)FE_ * FMID_), 256, 0, stream>>>(W2, W2T, FE_, FMID_);
  wtrans_cast_kernel<<<blocks((long)NH_ * FV_), 256, 0, stream>>>(W3, W3T, NH_, FV_);

  // ---- layer 1 (node): X1 = relu(((eye+(1-eye)*T s1 T^T)*adj_v) @ X W1 + b1)
  rowdot_kernel<<<(B_ * NE_ + 3) / 4, dim3(32, 4), 0, stream>>>(Z, p1, s1, B_ * NE_, FE_);
  { long n = (long)B_ * NV_ * NE_;
    scale_cast_f32_kernel<<<blocks(n), 256, 0, stream>>>(T, s1, Ts, NV_, NE_, n); }
  gemm_hwt_kernel<<<dim3(NV_ / 16, NH_ / 16, B_), 32, 0, stream>>>(
      Xbf, W1T, nullptr, HWT1, NV_, FV_, NH_);
  node_fused_kernel<NH_><<<dim3(NV_ / 16, B_), 32, 0, stream>>>(
      Ts, Tbf, adj_v, HWT1, b1, X1, X1bf, 1);

  // ---- layer 2 (edge): Z1 = relu((A/colmax) @ Z W2 + b2)
  rowdot_kernel<<<(B_ * NV_ + 3) / 4, dim3(32, 4), 0, stream>>>(X1, p2, s2, B_ * NV_, NH_);
  { long n = (long)B_ * NE_ * NV_;
    scale_cast_bf16_kernel<<<blocks(n), 256, 0, stream>>>(Ttbf, s2, Ts, NE_, NV_, n); }
  fill_kernel<<<blocks((long)B_ * NE_), 256, 0, stream>>>(cmax, -FLT_MAX, B_ * NE_);
  edge_pass1_kernel<<<dim3(NE_ / 16, B_), 32, 0, stream>>>(Ts, Ttbf, adj_e, Aedge, cmax);
  gemm_hwt_kernel<<<dim3(NE_ / 16, FMID_ / 16, B_), 32, 0, stream>>>(
      Zbf, W2T, cmax, HWT2, NE_, FE_, FMID_);  // 1/colmax folded into HW rows
  amatmul_kernel<FMID_><<<dim3(NE_ / 16, B_), 32, 0, stream>>>(
      Aedge, HWT2, b2, Z1, NE_, NE_, 1);

  // ---- layer 3 (node): out = ((eye+(1-eye)*T s3 T^T)*adj_v) @ X1 W3 + b3
  rowdot_kernel<<<(B_ * NE_ + 3) / 4, dim3(32, 4), 0, stream>>>(Z1, p3, s3, B_ * NE_, FMID_);
  { long n = (long)B_ * NV_ * NE_;
    scale_cast_f32_kernel<<<blocks(n), 256, 0, stream>>>(T, s3, Ts, NV_, NE_, n); }
  gemm_hwt_kernel<<<dim3(NV_ / 16, FV_ / 16, B_), 32, 0, stream>>>(
      X1bf, W3T, nullptr, HWT3, NV_, NH_, FV_);
  node_fused_kernel<FV_><<<dim3(NV_ / 16, B_), 32, 0, stream>>>(
      Ts, Tbf, adj_v, HWT3, b3, out, nullptr, 0);
}
